// EquivariantDenseAttention_19396072309328
// MI455X (gfx1250) — matrix operand order repaired and checked by reading, not compiled, not run on hardware
//
#include <hip/hip_runtime.h>

#define NQ 4096
#define NHEADS 8

typedef __bf16 bf16_t;
typedef bf16_t v16bf __attribute__((ext_vector_type(16)));
typedef float  v8f   __attribute__((ext_vector_type(8)));

struct __align__(16) U128 { unsigned int x, y, z, w; };

union Frag {
    v16bf v;
    U128 q[2];
};

__device__ __forceinline__ v8f zero8() {
    v8f z;
#pragma unroll
    for (int i = 0; i < 8; ++i) z[i] = 0.f;
    return z;
}

// ---------------- Kernel 0: w2 (f32) -> bf16 workspace copy -----------------
__global__ __launch_bounds__(256) void k0_w2bf16(const float* __restrict__ w2,
                                                 bf16_t* __restrict__ W2b,
                                                 int n) {
    int i = blockIdx.x * 256 + threadIdx.x;
    if (i < n) W2b[i] = (bf16_t)w2[i];
}

// ---------------- Kernel 1: radial MLP + low-rank TP -> Q/K/V ---------------
// One block = 16 nodes. C(16x192) = Z(16x8192) @ W2v(192x8192)^T, Z built on
// the fly from the rank-1 products t64[m,col]*h[m,k]. Then kqv = res x b2,
// split into K/Q/V bf16 (Q pre-scaled, V transposed).
__global__ __launch_bounds__(128) void k1_radial(
    const float* __restrict__ b1g, const float* __restrict__ b2g,
    const float* __restrict__ efg, const float* __restrict__ fg,
    const float* __restrict__ w1,  const float* __restrict__ bias1,
    const float* __restrict__ bias2, const int* __restrict__ src_idx,
    const bf16_t* __restrict__ W2b,
    bf16_t* __restrict__ Qb, bf16_t* __restrict__ Kb,
    bf16_t* __restrict__ Vt) {
    __shared__ float efs[16][64];     // edge feats tile
    __shared__ float hs[16][128];     // relu MLP hidden
    __shared__ float ts[16][64];      // t64 vectors
    __shared__ float res[16][200];    // 16x192 result (+pad)

    const int tid = threadIdx.x;
    const int n0  = blockIdx.x * 16;

    // stage edge features
#pragma unroll
    for (int i = 0; i < 8; ++i) {
        int idx = tid + i * 128;                       // 0..1023
        efs[idx >> 6][idx & 63] = efg[n0 * 64 + idx];
    }
    // t64[m][mm*2+r] = sum_d f[src,mm,d] * b1[n,d,r]
#pragma unroll
    for (int i = 0; i < 8; ++i) {
        int idx = tid + i * 128;                       // 0..1023
        int m = idx >> 6, c = idx & 63;
        int mm = c >> 1, r = c & 1;
        int n = n0 + m;
        int s = src_idx[n];
        float acc = 0.f;
#pragma unroll
        for (int d = 0; d < 4; ++d)
            acc += fg[s * 128 + mm * 4 + d] * b1g[n * 8 + d * 2 + r];
        ts[m][c] = acc;
    }
    __syncthreads();

    // h = relu(ef @ w1^T + bias1); thread = hidden unit k
    {
        int k = tid;
        float bk = bias1[k];
        for (int m = 0; m < 16; ++m) {
            float acc = bk;
#pragma unroll 8
            for (int e = 0; e < 64; ++e) acc += efs[m][e] * w1[k * 64 + e];
            hs[m][k] = fmaxf(acc, 0.f);
        }
    }
    __syncthreads();

    // bias2 contribution: res[m][x] = sum_col bias2[x*64+col] * t64[m][col]
#pragma unroll
    for (int i = 0; i < 24; ++i) {
        int idx = tid + i * 128;                       // 0..3071
        int m = idx & 15, x = idx >> 4;
        float acc = 0.f;
        for (int c = 0; c < 64; ++c) acc += bias2[x * 64 + c] * ts[m][c];
        res[m][x] = acc;
    }
    __syncthreads();

    // big GEMM: wave w owns x-tiles {3w, 3w+1, 3w+2}
    const int lane = tid & 31, wid = tid >> 5;
    const int half = lane >> 4, mrow = lane & 15;
    const int xb = wid * 48;
    v8f D0 = zero8(), D1 = zero8(), D2 = zero8();

    for (int col = 0; col < 64; ++col) {
        float t = ts[mrow][col];
#pragma unroll
        for (int kc = 0; kc < 4; ++kc) {
            // A fragment: z[mrow, col*128 + kc*32 + Kmap(i)], Kmap is this code's
            // contiguous order (B matches): lanes<16 -> K {0..15}, lanes>=16 -> {16..31}
            const float* hp = &hs[mrow][kc * 32 + half * 16];
            Frag A;
#pragma unroll
            for (int i = 0; i < 8; ++i) {
                A.v[i]     = (bf16_t)(t * hp[i]);
                A.v[i + 8] = (bf16_t)(t * hp[i + 8]);
            }
            // B fragments: lane = column x, 16 consecutive K along w2 row
            int jb = col * 128 + kc * 32 + half * 16;
            const bf16_t* wp0 = W2b + (unsigned)(xb + mrow)      * 8192u + jb;
            const bf16_t* wp1 = W2b + (unsigned)(xb + 16 + mrow) * 8192u + jb;
            const bf16_t* wp2 = W2b + (unsigned)(xb + 32 + mrow) * 8192u + jb;
            Frag B0, B1, B2;
            B0.q[0] = *(const U128*)wp0;  B0.q[1] = *(const U128*)(wp0 + 8);
            B1.q[0] = *(const U128*)wp1;  B1.q[1] = *(const U128*)(wp1 + 8);
            B2.q[0] = *(const U128*)wp2;  B2.q[1] = *(const U128*)(wp2 + 8);
            D0 = __builtin_amdgcn_wmma_f32_16x16x32_bf16(false, A.v, false, B0.v,
                                                         (short)0, D0, false, false);
            D1 = __builtin_amdgcn_wmma_f32_16x16x32_bf16(false, A.v, false, B1.v,
                                                         (short)0, D1, false, false);
            D2 = __builtin_amdgcn_wmma_f32_16x16x32_bf16(false, A.v, false, B2.v,
                                                         (short)0, D2, false, false);
        }
    }
    // accumulate into res (bias term already there); unique owner per element
#pragma unroll
    for (int g = 0; g < 8; ++g) {
        int m = g + 8 * half;
        res[m][xb +      mrow] += D0[g];
        res[m][xb + 16 + mrow] += D1[g];
        res[m][xb + 32 + mrow] += D2[g];
    }
    __syncthreads();

    // kqv[n,xx,d] = res[.,2xx]*b2[n,0,d] + res[.,2xx+1]*b2[n,1,d];
    // split into K (xx<32), Q (32..63, pre-scaled by 1/sqrt(16)), V (64..95)
#pragma unroll
    for (int i = 0; i < 48; ++i) {
        int idx = tid + i * 128;                       // 0..6143
        int m = idx & 15;
        int rest = idx >> 4;
        int d = rest & 3;
        int xx = rest >> 2;                            // 0..95
        int n = n0 + m;
        float r0 = res[m][2 * xx], r1 = res[m][2 * xx + 1];
        float v = r0 * b2g[n * 8 + d] + r1 * b2g[n * 8 + 4 + d];
        int xm = xx & 31;
        int flat = xm * 4 + d;
        int hh = flat >> 4, j = flat & 15;
        if (xx < 32)       Kb[(hh * NQ + n) * 16 + j] = (bf16_t)v;
        else if (xx < 64)  Qb[(hh * NQ + n) * 16 + j] = (bf16_t)(v * 0.25f);
        else               Vt[(hh * 16 + j) * NQ + n] = (bf16_t)v;
    }
}

// ---------------- Kernel 2: flash attention, one wave per (head, q-tile) ----
__global__ __launch_bounds__(256) void k2_attn(
    const bf16_t* __restrict__ Qb, const bf16_t* __restrict__ Kb,
    const bf16_t* __restrict__ Vt, float* __restrict__ Aout) {
    __shared__ __align__(16) bf16_t Pbuf[8][16][32];

    const int tid = threadIdx.x, wid = tid >> 5, lane = tid & 31;
    const int gid = blockIdx.x * 8 + wid;              // 0..2047
    const int h = gid >> 8, qt = gid & 255;
    const int q0 = qt * 16;
    const int half = lane >> 4, mcol = lane & 15;

    // Q fragment (head_dim 16 zero-padded to K=32)
    Frag Qf;
    const bf16_t* qp = Qb + (unsigned)(h * NQ + q0 + mcol) * 16u;
    Qf.q[0] = half ? U128{0, 0, 0, 0} : *(const U128*)qp;
    Qf.q[1] = half ? U128{0, 0, 0, 0} : *(const U128*)(qp + 8);

    v8f acc = zero8();
    float rm[8], rs[8];
#pragma unroll
    for (int g = 0; g < 8; ++g) { rm[g] = -3.0e38f; rs[g] = 0.f; }

    for (int kb = 0; kb < NQ; kb += 32) {
#pragma unroll
        for (int sub = 0; sub < 2; ++sub) {
            int kt = kb + sub * 16;
            // B = K^T (dims x keys): lanes>=16 carry the zero-padded dims
            Frag Kf;
            if (half == 0) {
                const bf16_t* kp = Kb + (unsigned)(h * NQ + kt + mcol) * 16u;
                Kf.q[0] = *(const U128*)kp;
                Kf.q[1] = *(const U128*)(kp + 8);
            } else {
                Kf.q[0] = U128{0, 0, 0, 0};
                Kf.q[1] = U128{0, 0, 0, 0};
            }
            v8f S = __builtin_amdgcn_wmma_f32_16x16x32_bf16(false, Qf.v, false, Kf.v,
                                                            (short)0, zero8(),
                                                            false, false);
            // online softmax; C layout: row = g + 8*half, col = mcol
#pragma unroll
            for (int g = 0; g < 8; ++g) {
                float v = S[g];
                float mx = v;
#pragma unroll
                for (int off = 8; off > 0; off >>= 1)
                    mx = fmaxf(mx, __shfl_xor(mx, off, 16));
                float mn = fmaxf(rm[g], mx);
                float corr = __expf(rm[g] - mn);
                float p = __expf(v - mn);
                float ps = p;
#pragma unroll
                for (int off = 8; off > 0; off >>= 1)
                    ps += __shfl_xor(ps, off, 16);
                rs[g] = rs[g] * corr + ps;
                acc[g] *= corr;
                rm[g] = mn;
                Pbuf[wid][g + 8 * half][sub * 16 + mcol] = (bf16_t)p;
            }
        }
        // P: C layout -> A layout via per-wave LDS staging
        Frag Pf;
        const bf16_t* pp = &Pbuf[wid][mcol][half * 16];
        Pf.q[0] = *(const U128*)pp;
        Pf.q[1] = *(const U128*)(pp + 8);
        // V fragment from transposed store: lane = dim, 16 consecutive keys
        const bf16_t* vp = Vt + (unsigned)(h * 16 + mcol) * (unsigned)NQ
                               + kb + half * 16;
        Frag Vf;
        Vf.q[0] = *(const U128*)vp;
        Vf.q[1] = *(const U128*)(vp + 8);
        acc = __builtin_amdgcn_wmma_f32_16x16x32_bf16(false, Pf.v, false, Vf.v,
                                                      (short)0, acc, false, false);
    }
#pragma unroll
    for (int g = 0; g < 8; ++g)
        Aout[(q0 + g + 8 * half) * 128 + h * 16 + mcol] = acc[g] / rs[g];
}

// ---------------- Kernel 3: equivariant linear projection -------------------
__global__ __launch_bounds__(256) void k3_proj(
    const float* __restrict__ Aout, const float* __restrict__ pw,
    const float* __restrict__ pb, float* __restrict__ out) {
    int id = blockIdx.x * 256 + threadIdx.x;           // 0..131071
    int n = id >> 5, m = id & 31;
    float a0 = pb[m], a1 = 0.f, a2 = 0.f, a3 = 0.f;
    const float* ar = Aout + n * 128;
#pragma unroll 4
    for (int mp = 0; mp < 32; ++mp) {
        float w0 = pw[m * 32 + mp];                    // l=0 block (scalar slot)
        float w1 = pw[(m + 32) * 32 + mp];             // l=1 block (d=1..3)
        a0 += w0 * ar[mp * 4 + 0];
        a1 += w1 * ar[mp * 4 + 1];
        a2 += w1 * ar[mp * 4 + 2];
        a3 += w1 * ar[mp * 4 + 3];
    }
    float* op = out + n * 128 + m * 4;
    op[0] = a0; op[1] = a1; op[2] = a2; op[3] = a3;
}

extern "C" void kernel_launch(void* const* d_in, const int* in_sizes, int n_in,
                              void* d_out, int out_size, void* d_ws, size_t ws_size,
                              hipStream_t stream) {
    (void)in_sizes; (void)n_in; (void)out_size; (void)ws_size;
    const float* b1     = (const float*)d_in[0];
    const float* b2     = (const float*)d_in[1];
    const float* ef     = (const float*)d_in[2];
    const float* f      = (const float*)d_in[3];
    const float* w1     = (const float*)d_in[4];
    const float* bias1  = (const float*)d_in[5];
    const float* w2     = (const float*)d_in[6];
    const float* bias2  = (const float*)d_in[7];
    const float* proj_w = (const float*)d_in[8];
    const float* proj_b = (const float*)d_in[9];
    const int*   srcix  = (const int*)d_in[10];

    char* ws = (char*)d_ws;
    bf16_t* W2b = (bf16_t*)(ws);              // 3,145,728 B
    bf16_t* Qb  = (bf16_t*)(ws + 3145728);    // 1 MB
    bf16_t* Kb  = (bf16_t*)(ws + 4194304);    // 1 MB
    bf16_t* Vt  = (bf16_t*)(ws + 5242880);    // 1 MB
    float*  Ao  = (float*)(ws + 6291456);     // 2 MB

    k0_w2bf16<<<6144, 256, 0, stream>>>(w2, W2b, 12288 * 128);
    k1_radial<<<256, 128, 0, stream>>>(b1, b2, ef, f, w1, bias1, bias2, srcix,
                                       W2b, Qb, Kb, Vt);
    k2_attn<<<256, 256, 0, stream>>>(Qb, Kb, Vt, Ao);
    k3_proj<<<512, 256, 0, stream>>>(Ao, proj_w, proj_b, (float*)d_out);
}
